// HeteroGNN_12017318494617
// MI455X (gfx1250) — matrix operand order repaired and checked by reading, compiled, not run on hardware
//
#include <hip/hip_runtime.h>
#include <hip/hip_bf16.h>

// ---------------------------------------------------------------------------
// HeteroGNN (2-layer SAGE) forward for MI455X / gfx1250.
// Dense math: v_wmma_f32_16x16x32_bf16 (f32 accumulate), software-pipelined.
// Sparse aggregation: edge-parallel global_atomic_add_f32 with bf16 gathers.
// ---------------------------------------------------------------------------

#define HID 128
#define OUTD 64
#define FM  404
#define FMP 416   // 404 padded to multiple of 32
#define FU  16
#define FUP 32    // 16 padded to 32
#define NU  100000
#define NM  50000
#define NE  1000000

typedef __attribute__((ext_vector_type(16))) __bf16 v16bf;
typedef __attribute__((ext_vector_type(8)))  __bf16 v8bf;
typedef __attribute__((ext_vector_type(4)))  __bf16 v4bf;
typedef __attribute__((ext_vector_type(8)))  float  v8f;

// ---------------- prep kernels ----------------------------------------------

// row-major f32 [rows, sc] -> bf16 [rows, dc] zero-padded
__global__ void __launch_bounds__(256)
cvt_pad_bf16(const float* __restrict__ src, __bf16* __restrict__ dst,
             int rows, int sc, int dc) {
  long long i = (long long)blockIdx.x * 256 + threadIdx.x;
  long long total = (long long)rows * dc;
  if (i >= total) return;
  int r = (int)(i / dc), c = (int)(i - (long long)r * dc);
  dst[i] = (c < sc) ? (__bf16)src[(long long)r * sc + c] : (__bf16)0.0f;
}

// f32 W[K,N] -> bf16 Wt[N,Kpad] (transposed, K zero-padded)
__global__ void __launch_bounds__(256)
wt_transpose_bf16(const float* __restrict__ w, __bf16* __restrict__ wt,
                  int K, int N, int Kpad) {
  long long i = (long long)blockIdx.x * 256 + threadIdx.x;
  long long total = (long long)N * Kpad;
  if (i >= total) return;
  int n = (int)(i / Kpad), k = (int)(i - (long long)n * Kpad);
  wt[i] = (k < K) ? (__bf16)w[(long long)k * N + n] : (__bf16)0.0f;
}

// ---------------- graph kernels ---------------------------------------------

__global__ void __launch_bounds__(256)
deg_count(const int* __restrict__ src, const int* __restrict__ dst,
          float* __restrict__ deg_u, float* __restrict__ deg_m, int E) {
  int e = blockIdx.x * 256 + threadIdx.x;
  if (e >= E) return;
  atomicAdd(&deg_u[src[e]], 1.0f);
  atomicAdd(&deg_m[dst[e]], 1.0f);
}

// one wave (32 lanes) per edge; 128 channels = 4 bf16 per lane
__global__ void __launch_bounds__(256)
scatter_add_bf16(const __bf16* __restrict__ X, const int* __restrict__ from_i,
                 const int* __restrict__ to_i, float* __restrict__ sum, int E) {
  int gw = (blockIdx.x * 256 + threadIdx.x) >> 5;
  int lane = threadIdx.x & 31;
  if (gw >= E) return;
  int a = __builtin_amdgcn_readfirstlane(from_i[gw]);   // wave-uniform -> SGPR base
  int b = __builtin_amdgcn_readfirstlane(to_i[gw]);
  const v4bf v = *(const v4bf*)(X + (long long)a * HID + lane * 4);
  float* dst = sum + (long long)b * HID + lane * 4;
#pragma unroll
  for (int j = 0; j < 4; ++j) atomicAdd(dst + j, (float)v[j]);
}

// mean = sum / max(deg,1) -> bf16
__global__ void __launch_bounds__(256)
mean_to_bf16(const float* __restrict__ sum, const float* __restrict__ deg,
             __bf16* __restrict__ out, long long total) {
  long long i = (long long)blockIdx.x * 256 + threadIdx.x;
  if (i >= total) return;
  float d = deg[i >> 7];             // D == 128
  d = d < 1.0f ? 1.0f : d;
  out[i] = (__bf16)(sum[i] / d);
}

// ---------------- WMMA fragment loaders (ISA 7.12.2 layouts) ----------------

// A 16x32 bf16: lane holds row m=lane%16; elems 0..7 = K kb+half*8+0..7,
// elems 8..15 = K kb+16+half*8+0..7  (half = lane/16)
__device__ inline v16bf load_a_frag(const __bf16* A, int lda, int m0,
                                    int lane, int kb) {
  int m = m0 + (lane & 15);
  int k = kb + ((lane >> 4) << 3);
  const __bf16* p = A + (long long)m * lda + k;
  v8bf lo = *(const v8bf*)p;
  v8bf hi = *(const v8bf*)(p + 16);
  v16bf r;
#pragma unroll
  for (int i = 0; i < 8; ++i) { r[i] = lo[i]; r[i + 8] = hi[i]; }
  return r;
}

// B 32x16 bf16 from transposed weights Wt[N,Kpad]: lane holds col n=lane%16,
// elems 0..7 = K kb+half*8+0..7, elems 8..15 = K kb+16+half*8+0..7
__device__ inline v16bf load_b_frag(const __bf16* Wt, int Kpad, int n0,
                                    int lane, int kb) {
  int n = n0 + (lane & 15);
  int k = kb + ((lane >> 4) << 3);
  const __bf16* p = Wt + (long long)n * Kpad + k;
  v8bf lo = *(const v8bf*)p;
  v8bf hi = *(const v8bf*)(p + 16);
  v16bf r;
#pragma unroll
  for (int i = 0; i < 8; ++i) { r[i] = lo[i]; r[i + 8] = hi[i]; }
  return r;
}

// ---------------- fused GEMM + bias + (ReLU) + LayerNorm + (ReLU) -----------
// Y[M,N] = LN( act( A1@W1t^T + (HAS_A2? A2@W2t^T : 0) + bias ) )
// Block: 16 rows x N cols; one wave per 16-col tile; blockDim = 2*N threads.
// K-loop is software pipelined (load kb+32 before WMMA on kb), branch-free.
template <bool HAS_A2>
__global__ void __launch_bounds__(256)
sage_gemm_ln(const __bf16* __restrict__ A1, const __bf16* __restrict__ W1t,
             const __bf16* __restrict__ A2, const __bf16* __restrict__ W2t,
             const float* __restrict__ bias, const float* __restrict__ gamma,
             const float* __restrict__ beta,
             float* __restrict__ outF, __bf16* __restrict__ outB,
             int N, int K, int preRelu, int postRelu) {
  __shared__ float tile[16 * HID];
  __shared__ float red_s[256], red_ss[256];
  __shared__ float mu_s[16], rs_s[16];

  const int tid  = threadIdx.x;
  const int lane = tid & 31;
  const int wave = tid >> 5;           // 0 .. N/16-1
  const int n0   = wave * 16;
  const int m0   = blockIdx.x * 16;

  v8f acc = {};
  // prologue fragments
  v16bf a1 = load_a_frag(A1, K, m0, lane, 0);
  v16bf b1 = load_b_frag(W1t, K, n0, lane, 0);
  v16bf a2, b2;
  if (HAS_A2) {
    a2 = load_a_frag(A2, K, m0, lane, 0);
    b2 = load_b_frag(W2t, K, n0, lane, 0);
  }
  // steady state: prefetch kb+32, WMMA on kb
  for (int kb = 0; kb + 32 < K; kb += 32) {
    v16bf a1n = load_a_frag(A1, K, m0, lane, kb + 32);
    v16bf b1n = load_b_frag(W1t, K, n0, lane, kb + 32);
    v16bf a2n, b2n;
    if (HAS_A2) {
      a2n = load_a_frag(A2, K, m0, lane, kb + 32);
      b2n = load_b_frag(W2t, K, n0, lane, kb + 32);
    }
    acc = __builtin_amdgcn_wmma_f32_16x16x32_bf16(
        false, a1, false, b1, (short)0, acc, false, false);
    if (HAS_A2)
      acc = __builtin_amdgcn_wmma_f32_16x16x32_bf16(
          false, a2, false, b2, (short)0, acc, false, false);
    a1 = a1n; b1 = b1n;
    if (HAS_A2) { a2 = a2n; b2 = b2n; }
  }
  // epilogue (last K step)
  acc = __builtin_amdgcn_wmma_f32_16x16x32_bf16(
      false, a1, false, b1, (short)0, acc, false, false);
  if (HAS_A2)
    acc = __builtin_amdgcn_wmma_f32_16x16x32_bf16(
        false, a2, false, b2, (short)0, acc, false, false);

  const float bv = bias[n0 + (lane & 15)];
#pragma unroll
  for (int r = 0; r < 8; ++r) {                 // C layout: row r + 8*(lane/16)
    int ml = r + ((lane >> 4) << 3);
    float v = acc[r] + bv;
    if (preRelu) v = fmaxf(v, 0.0f);
    tile[ml * N + n0 + (lane & 15)] = v;
  }
  __syncthreads();

  // parallel LN stats: blockDim/16 threads per row -> partials -> fold
  const int tpr = blockDim.x >> 4;              // threads per row (16 or 8)
  {
    int r = tid / tpr, j = tid - r * tpr;
    float s = 0.0f, ss = 0.0f;
    const float* row = &tile[r * N];
    for (int c = j; c < N; c += tpr) { float v = row[c]; s += v; ss += v * v; }
    red_s[tid] = s;
    red_ss[tid] = ss;
  }
  __syncthreads();
  if (tid < 16) {
    float s = 0.0f, ss = 0.0f;
    for (int j = 0; j < tpr; ++j) { s += red_s[tid * tpr + j]; ss += red_ss[tid * tpr + j]; }
    float mu  = s / (float)N;
    float var = ss / (float)N - mu * mu;
    mu_s[tid] = mu;
    rs_s[tid] = rsqrtf(var + 1e-5f);
  }
  __syncthreads();

  for (int t = tid; t < 16 * N; t += blockDim.x) {
    int r = t / N, c = t - r * N;
    float y = (tile[t] - mu_s[r]) * rs_s[r] * gamma[c] + beta[c];
    if (postRelu) y = fmaxf(y, 0.0f);
    long long o = (long long)(m0 + r) * N + c;
    if (outF) outF[o] = y;
    if (outB) outB[o] = (__bf16)y;
  }
}

// ---------------------------------------------------------------------------

static inline long long cdiv(long long a, long long b) { return (a + b - 1) / b; }
static inline size_t alignup(size_t x) { return (x + 255) & ~(size_t)255; }

extern "C" void kernel_launch(void* const* d_in, const int* in_sizes, int n_in,
                              void* d_out, int out_size, void* d_ws, size_t ws_size,
                              hipStream_t stream) {
  (void)in_sizes; (void)n_in; (void)out_size; (void)ws_size;
  // inputs (setup_inputs order)
  const float* x_user  = (const float*)d_in[0];
  const float* x_movie = (const float*)d_in[1];
  const int*   e_src   = (const int*)d_in[2];   // user ids
  const int*   e_dst   = (const int*)d_in[3];   // movie ids
  const float* enc_u_w = (const float*)d_in[4];
  const float* enc_u_b = (const float*)d_in[5];
  const float* enc_u_g = (const float*)d_in[6];
  const float* enc_u_be= (const float*)d_in[7];
  const float* enc_m_w = (const float*)d_in[8];
  const float* enc_m_b = (const float*)d_in[9];
  const float* enc_m_g = (const float*)d_in[10];
  const float* enc_m_be= (const float*)d_in[11];
  const float* c1m_wl  = (const float*)d_in[12];
  const float* c1m_bl  = (const float*)d_in[13];
  const float* c1m_wr  = (const float*)d_in[14];
  const float* c1u_wl  = (const float*)d_in[15];
  const float* c1u_bl  = (const float*)d_in[16];
  const float* c1u_wr  = (const float*)d_in[17];
  const float* ln1_u_g = (const float*)d_in[18];
  const float* ln1_u_b = (const float*)d_in[19];
  const float* ln1_m_g = (const float*)d_in[20];
  const float* ln1_m_b = (const float*)d_in[21];
  const float* c2m_wl  = (const float*)d_in[22];
  const float* c2m_bl  = (const float*)d_in[23];
  const float* c2m_wr  = (const float*)d_in[24];
  const float* c2u_wl  = (const float*)d_in[25];
  const float* c2u_bl  = (const float*)d_in[26];
  const float* c2u_wr  = (const float*)d_in[27];
  const float* ln2_u_g = (const float*)d_in[28];
  const float* ln2_u_b = (const float*)d_in[29];
  const float* ln2_m_g = (const float*)d_in[30];
  const float* ln2_m_b = (const float*)d_in[31];

  // workspace layout
  char* p = (char*)d_ws;
  auto take = [&](size_t bytes) { char* q = p; p += alignup(bytes); return q; };
  float*  sum_m   = (float*) take((size_t)NM * HID * 4);
  float*  sum_u   = (float*) take((size_t)NU * HID * 4);
  float*  deg_m   = (float*) take((size_t)NM * 4);
  float*  deg_u   = (float*) take((size_t)NU * 4);
  __bf16* xu_bf   = (__bf16*)take((size_t)NU * FUP * 2);
  __bf16* xm_bf   = (__bf16*)take((size_t)NM * FMP * 2);
  __bf16* hu_bf   = (__bf16*)take((size_t)NU * HID * 2);
  __bf16* hm_bf   = (__bf16*)take((size_t)NM * HID * 2);
  __bf16* u1_bf   = (__bf16*)take((size_t)NU * HID * 2);
  __bf16* m1_bf   = (__bf16*)take((size_t)NM * HID * 2);
  __bf16* mean_u  = (__bf16*)take((size_t)NU * HID * 2);
  __bf16* mean_m  = (__bf16*)take((size_t)NM * HID * 2);
  __bf16* wt_eu   = (__bf16*)take((size_t)HID * FUP * 2);
  __bf16* wt_em   = (__bf16*)take((size_t)HID * FMP * 2);
  __bf16* wt_1mwl = (__bf16*)take((size_t)HID * HID * 2);
  __bf16* wt_1mwr = (__bf16*)take((size_t)HID * HID * 2);
  __bf16* wt_1uwl = (__bf16*)take((size_t)HID * HID * 2);
  __bf16* wt_1uwr = (__bf16*)take((size_t)HID * HID * 2);
  __bf16* wt_2mwl = (__bf16*)take((size_t)OUTD * HID * 2);
  __bf16* wt_2mwr = (__bf16*)take((size_t)OUTD * HID * 2);
  __bf16* wt_2uwl = (__bf16*)take((size_t)OUTD * HID * 2);
  __bf16* wt_2uwr = (__bf16*)take((size_t)OUTD * HID * 2);

  float* out_u2 = (float*)d_out;                        // [NU, 64]
  float* out_m2 = (float*)d_out + (size_t)NU * OUTD;    // [NM, 64]

  // zero accumulators / degrees
  hipMemsetAsync(sum_m, 0, (size_t)NM * HID * 4, stream);
  hipMemsetAsync(sum_u, 0, (size_t)NU * HID * 4, stream);
  hipMemsetAsync(deg_m, 0, (size_t)NM * 4, stream);
  hipMemsetAsync(deg_u, 0, (size_t)NU * 4, stream);

  // bf16 activation conversion (padded K)
  cvt_pad_bf16<<<(int)cdiv((long long)NU * FUP, 256), 256, 0, stream>>>(
      x_user, xu_bf, NU, FU, FUP);
  cvt_pad_bf16<<<(int)cdiv((long long)NM * FMP, 256), 256, 0, stream>>>(
      x_movie, xm_bf, NM, FM, FMP);

  // bf16 transposed weights [N, Kpad]
  wt_transpose_bf16<<<(int)cdiv((long long)HID * FUP, 256), 256, 0, stream>>>(enc_u_w, wt_eu, FU, HID, FUP);
  wt_transpose_bf16<<<(int)cdiv((long long)HID * FMP, 256), 256, 0, stream>>>(enc_m_w, wt_em, FM, HID, FMP);
  wt_transpose_bf16<<<(int)cdiv((long long)HID * HID, 256), 256, 0, stream>>>(c1m_wl, wt_1mwl, HID, HID, HID);
  wt_transpose_bf16<<<(int)cdiv((long long)HID * HID, 256), 256, 0, stream>>>(c1m_wr, wt_1mwr, HID, HID, HID);
  wt_transpose_bf16<<<(int)cdiv((long long)HID * HID, 256), 256, 0, stream>>>(c1u_wl, wt_1uwl, HID, HID, HID);
  wt_transpose_bf16<<<(int)cdiv((long long)HID * HID, 256), 256, 0, stream>>>(c1u_wr, wt_1uwr, HID, HID, HID);
  wt_transpose_bf16<<<(int)cdiv((long long)OUTD * HID, 256), 256, 0, stream>>>(c2m_wl, wt_2mwl, HID, OUTD, HID);
  wt_transpose_bf16<<<(int)cdiv((long long)OUTD * HID, 256), 256, 0, stream>>>(c2m_wr, wt_2mwr, HID, OUTD, HID);
  wt_transpose_bf16<<<(int)cdiv((long long)OUTD * HID, 256), 256, 0, stream>>>(c2u_wl, wt_2uwl, HID, OUTD, HID);
  wt_transpose_bf16<<<(int)cdiv((long long)OUTD * HID, 256), 256, 0, stream>>>(c2u_wr, wt_2uwr, HID, OUTD, HID);

  // degrees (shared by both layers)
  deg_count<<<(int)cdiv(NE, 256), 256, 0, stream>>>(e_src, e_dst, deg_u, deg_m, NE);

  // encoders: hu = LN(relu(x_user@Wu + b)); hm = LN(relu(x_movie@Wm + b))
  sage_gemm_ln<false><<<NU / 16, 2 * HID, 0, stream>>>(
      xu_bf, wt_eu, nullptr, nullptr, enc_u_b, enc_u_g, enc_u_be,
      nullptr, hu_bf, HID, FUP, 1, 0);
  sage_gemm_ln<false><<<NM / 16, 2 * HID, 0, stream>>>(
      xm_bf, wt_em, nullptr, nullptr, enc_m_b, enc_m_g, enc_m_be,
      nullptr, hm_bf, HID, FMP, 1, 0);

  // layer-1 aggregation: sum_m[dst] += hu[src]; sum_u[src] += hm[dst]
  const int scat_blocks = (int)cdiv((long long)NE * 32, 256);
  scatter_add_bf16<<<scat_blocks, 256, 0, stream>>>(hu_bf, e_src, e_dst, sum_m, NE);
  scatter_add_bf16<<<scat_blocks, 256, 0, stream>>>(hm_bf, e_dst, e_src, sum_u, NE);
  mean_to_bf16<<<(int)cdiv((long long)NM * HID, 256), 256, 0, stream>>>(sum_m, deg_m, mean_m, (long long)NM * HID);
  mean_to_bf16<<<(int)cdiv((long long)NU * HID, 256), 256, 0, stream>>>(sum_u, deg_u, mean_u, (long long)NU * HID);

  // layer 1: relu(LN(mean@wl + bl + x@wr))
  sage_gemm_ln<true><<<NM / 16, 2 * HID, 0, stream>>>(
      mean_m, wt_1mwl, hm_bf, wt_1mwr, c1m_bl, ln1_m_g, ln1_m_b,
      nullptr, m1_bf, HID, HID, 0, 1);
  sage_gemm_ln<true><<<NU / 16, 2 * HID, 0, stream>>>(
      mean_u, wt_1uwl, hu_bf, wt_1uwr, c1u_bl, ln1_u_g, ln1_u_b,
      nullptr, u1_bf, HID, HID, 0, 1);

  // layer-2 aggregation
  hipMemsetAsync(sum_m, 0, (size_t)NM * HID * 4, stream);
  hipMemsetAsync(sum_u, 0, (size_t)NU * HID * 4, stream);
  scatter_add_bf16<<<scat_blocks, 256, 0, stream>>>(u1_bf, e_src, e_dst, sum_m, NE);
  scatter_add_bf16<<<scat_blocks, 256, 0, stream>>>(m1_bf, e_dst, e_src, sum_u, NE);
  mean_to_bf16<<<(int)cdiv((long long)NM * HID, 256), 256, 0, stream>>>(sum_m, deg_m, mean_m, (long long)NM * HID);
  mean_to_bf16<<<(int)cdiv((long long)NU * HID, 256), 256, 0, stream>>>(sum_u, deg_u, mean_u, (long long)NU * HID);

  // layer 2: LN(mean@wl + bl + x@wr) -> d_out (u2 first, then m2)
  sage_gemm_ln<true><<<NM / 16, 2 * OUTD, 0, stream>>>(
      mean_m, wt_2mwl, m1_bf, wt_2mwr, c2m_bl, ln2_m_g, ln2_m_b,
      out_m2, nullptr, OUTD, HID, 0, 0);
  sage_gemm_ln<true><<<NU / 16, 2 * OUTD, 0, stream>>>(
      mean_u, wt_2uwl, u1_bf, wt_2uwr, c2u_bl, ln2_u_g, ln2_u_b,
      out_u2, nullptr, OUTD, HID, 0, 0);
}
